// TTMMoE_85203561218082
// MI455X (gfx1250) — matrix operand order repaired and checked by reading, not compiled, not run on hardware
//
#include <hip/hip_runtime.h>

typedef __attribute__((ext_vector_type(16))) __bf16 v16bf;
typedef __attribute__((ext_vector_type(8)))  __bf16 v8bf;
typedef __attribute__((ext_vector_type(8)))  float  v8f;

#define NTOK   16384      // B*T
#define KDIM   1024
#define ODIM   1024
#define N1     32
#define RR     64
#define MTILE  16
#define BN_EPS 1e-5f

// ---------------- workspace layout (bytes) ----------------
#define OFF_BN   0u                                   // 64 floats: sum[32], sumsq[32]
#define OFF_H    1024u                                // h [16384][32] f32  (2 MB)
#define OFF_A1   (OFF_H  + (unsigned)NTOK*N1*4u)      // a1 [16384][32] f32 (2 MB)
#define OFF_F1   (OFF_A1 + (unsigned)NTOK*N1*4u)      // F1bT [4096][1024] bf16 (8 MB)
#define OFF_B1   (OFF_F1 + 4096u*1024u*2u)            // Bias1 [4096] f32
#define OFF_F2   (OFF_B1 + 4096u*4u)                  // F2b [1024][4096] bf16 (8 MB)

// ---------------- dynamic LDS layout (bytes) ----------------
#define SM_A    0u          // A_s  [16][64][64] bf16  = 128 KB
#define SM_ZB   131072u     // ZB   [16][1024]   bf16  =  32 KB
#define SM_G    163840u     // G_s  [16][1024]   bf16  =  32 KB  (c = q*64+s)
#define SM_M    196608u     // M_s  [16][1024]   bf16  =  32 KB  (c = q*64+r)
#define SM_A1   229376u     // a1   [16][32]     f32   =   2 KB
#define SMEM_BYTES 231424u

// sched_group_barrier masks
#define SG_WMMA   0x008
#define SG_VMEMRD 0x020
#define SG_DSRD   0x100

// -------- WMMA helpers (wave32, 16x16x32 bf16) --------
__device__ __forceinline__ v8f wmma_bf16(v16bf a, v16bf b, v8f c) {
  return __builtin_amdgcn_wmma_f32_16x16x32_bf16(false, a, false, b, (short)0, c,
                                                 false, false);
}

// A-operand fragment: row = M-row base pointer (K-contiguous).
// lane_hi==0 -> K {kb+0..15}; lane_hi==1 -> K {kb+16..31}
__device__ __forceinline__ v16bf fragA(const __bf16* row, int kb, int lane_hi) {
  const __bf16* p = row + kb + lane_hi * 16;
  v8bf lo = *(const v8bf*)(p);
  v8bf hi = *(const v8bf*)(p + 8);
  v16bf r;
#pragma unroll
  for (int i = 0; i < 8; ++i) { r[i] = lo[i]; r[i + 8] = hi[i]; }
  return r;
}

// B-operand fragment: col = N-column base pointer (K-contiguous).
// lanes 0-15: K kb+0..15, lanes 16-31: K kb+16..31
// `off` should be a compile-time constant so it folds into the vmem offset.
__device__ __forceinline__ v16bf fragB(const __bf16* col, int off, int lane_hi) {
  return *(const v16bf*)(col + off + lane_hi * 16);
}

// ================= kernel 0: zero BN accumulators =================
__global__ void zero_bn_kernel(float* bn) { bn[threadIdx.x] = 0.0f; }

// ============== kernel 1: factor1 -> F1bT bf16 + bias ==============
// F1bT[C][x] = factor1[s][x][q],  C = q*64+s ;  Bias1[C] = factor1[s][1024][q]
__global__ void conv_f1_kernel(const float* __restrict__ f1,
                               __bf16* __restrict__ F1bT,
                               float* __restrict__ Bias1) {
  const int C = blockIdx.x;           // 0..4095
  const int s = C & 63, q = C >> 6;
  const int tid = threadIdx.x;
#pragma unroll
  for (int j = 0; j < 4; ++j) {
    const int x = tid + j * 256;
    F1bT[(size_t)C * 1024 + x] = (__bf16)f1[((size_t)s * 1025 + x) * 64 + q];
  }
  if (tid == 0) Bias1[C] = f1[((size_t)s * 1025 + 1024) * 64 + q];
}

// ============== kernel 2: factor2 -> F2b bf16 ==============
// F2b[o][Kc] = factor2[q][o][r],  Kc = q*64+r
__global__ void conv_f2_kernel(const float* __restrict__ f2,
                               __bf16* __restrict__ F2b) {
  const int o = blockIdx.x;           // 0..1023
  const int tid = threadIdx.x;
#pragma unroll
  for (int j = 0; j < 16; ++j) {
    const int Kc = tid + j * 256;
    const int q = Kc >> 6, r = Kc & 63;
    F2b[(size_t)o * 4096 + Kc] = (__bf16)f2[((size_t)q * ODIM + o) * 64 + r];
  }
}

// ============== kernel 3: gating GEMM + BN partial sums ==============
// block = 256 = 8 tokens x 32 channels (one wave per token)
__global__ void gating_kernel(const float* __restrict__ z,
                              const float* __restrict__ proj0,
                              float* __restrict__ h, float* __restrict__ bn) {
  __shared__ float s1[8][32], s2[8][32];
  const int tid = threadIdx.x;
  const int tl = tid >> 5, n = tid & 31;
  const int token = blockIdx.x * 8 + tl;
  const float* zr = z + (size_t)token * KDIM;
  float acc = 0.0f;
#pragma unroll 8
  for (int k = 0; k < KDIM; ++k) acc += zr[k] * proj0[k * N1 + n];
  h[(size_t)token * N1 + n] = acc;
  s1[tl][n] = acc;
  s2[tl][n] = acc * acc;
  __syncthreads();
  if (tid < 32) {
    float a = 0.0f, b = 0.0f;
#pragma unroll
    for (int j = 0; j < 8; ++j) { a += s1[j][tid]; b += s2[j][tid]; }
    atomicAdd(bn + tid, a);
    atomicAdd(bn + 32 + tid, b);
  }
}

// ============== kernel 4: BN normalize + exact 1.5-entmax ==============
__global__ void entmax_kernel(const float* __restrict__ h,
                              const float* __restrict__ bn,
                              float* __restrict__ a1f) {
  const int token = blockIdx.x * blockDim.x + threadIdx.x;
  const float inv = 1.0f / (float)NTOK;
  float x[32];
  float mx = -1e30f;
#pragma unroll
  for (int n = 0; n < N1; ++n) {
    const float mu = bn[n] * inv;
    const float var = bn[32 + n] * inv - mu * mu;
    float v = (h[(size_t)token * N1 + n] - mu) * rsqrtf(var + BN_EPS);
    x[n] = v;
    mx = fmaxf(mx, v);
  }
  float xs[32];
#pragma unroll
  for (int n = 0; n < N1; ++n) { x[n] = (x[n] - mx) * 0.5f; xs[n] = x[n]; }
  // insertion sort, descending
  for (int i = 1; i < N1; ++i) {
    const float key = xs[i];
    int j = i - 1;
    while (j >= 0 && xs[j] < key) { xs[j + 1] = xs[j]; --j; }
    xs[j + 1] = key;
  }
  float tau[32];
  float cs = 0.0f, cs2 = 0.0f;
#pragma unroll
  for (int j = 0; j < N1; ++j) {
    cs += xs[j]; cs2 += xs[j] * xs[j];
    const float rho = (float)(j + 1);
    const float mean = cs / rho, meansq = cs2 / rho;
    const float ss = rho * (meansq - mean * mean);
    const float delta = fmaxf((1.0f - ss) / rho, 0.0f);
    tau[j] = mean - sqrtf(delta);
  }
  int support = 0;
#pragma unroll
  for (int j = 0; j < N1; ++j) support += (tau[j] <= xs[j]) ? 1 : 0;
  const float ts = tau[support - 1];
#pragma unroll
  for (int n = 0; n < N1; ++n) {
    const float v = fmaxf(x[n] - ts, 0.0f);
    a1f[(size_t)token * N1 + n] = v * v;
  }
}

// ============== kernel 5: fused TT chain (all-WMMA, software-pipelined) ====
// grid = 1024 WGs (16 tokens each), 256 threads = 8 waves.
// LDS-bound to 1 WG/WGP -> 2 waves/SIMD; launch_bounds unlocks a 512-VGPR
// budget. Triple-buffered B fragments (prefetch distance 2) with
// sched_group_barrier directives pinning the per-step order:
//   2x ds_read (A frag) -> 16x vmem_read (B frags k+2) -> 8x wmma (step k)
__global__ void __launch_bounds__(256, 2)
tt_fused_kernel(const float* __restrict__ z,
                const float* __restrict__ a1f,
                const float* __restrict__ F0,
                const __bf16* __restrict__ F1bT,
                const float* __restrict__ Bias1,
                const __bf16* __restrict__ F2b,
                float* __restrict__ out) {
  extern __shared__ char smem[];
  __bf16* A_s = (__bf16*)(smem + SM_A);    // [16][64*64]  r*64+s
  __bf16* ZB  = (__bf16*)(smem + SM_ZB);   // [16][1024]
  __bf16* G_s = (__bf16*)(smem + SM_G);    // [16][1024]   c=q*64+s
  __bf16* M_s = (__bf16*)(smem + SM_M);    // [16][1024]   c=q*64+r
  float*  A1  = (float*)(smem + SM_A1);    // [16][32]

  const int tid = threadIdx.x;
  const int wave = tid >> 5, lane = tid & 31;
  const int lane_lo = lane & 15, lane_hi = lane >> 4;
  const int tok0 = blockIdx.x * MTILE;

  // ---- stage 0a: a1 tile -> LDS ----
  A1[tid]       = a1f[(size_t)tok0 * N1 + tid];
  A1[tid + 256] = a1f[(size_t)tok0 * N1 + tid + 256];

  // ---- stage 0b: zb tile f32 -> bf16 LDS ----
  {
    const int t = tid >> 4;
    const int c0 = (tid & 15) * 4;
#pragma unroll
    for (int j = 0; j < 16; ++j) {
      const int c = c0 + j * 64;
      const float4 v = *(const float4*)(z + (size_t)(tok0 + t) * KDIM + c);
      ZB[t * 1024 + c + 0] = (__bf16)v.x;
      ZB[t * 1024 + c + 1] = (__bf16)v.y;
      ZB[t * 1024 + c + 2] = (__bf16)v.z;
      ZB[t * 1024 + c + 3] = (__bf16)v.w;
    }
  }
  __syncthreads();

  // ---- stage 0c: A_t[r][s] = sum_x a1[t][x] * F0[r][x][s]  -> bf16 LDS ----
  for (int p0 = 0; p0 < 16; ++p0) {
    const int p = p0 * 256 + tid;       // p = r*64+s, coalesced over s
    const int r = p >> 6, s = p & 63;
    float acc[MTILE];
#pragma unroll
    for (int t = 0; t < MTILE; ++t) acc[t] = 0.0f;
    for (int xx = 0; xx < N1; ++xx) {
      const float f = F0[((size_t)r * N1 + xx) * 64 + s];
#pragma unroll
      for (int t = 0; t < MTILE; ++t) acc[t] += A1[t * N1 + xx] * f;
    }
#pragma unroll
    for (int t = 0; t < MTILE; ++t) A_s[t * 4096 + p] = (__bf16)acc[t];
  }
  __syncthreads();

  // persistent GEMM2 accumulators: wave owns out columns [wave*128, wave*128+128)
  v8f outacc[8];
#pragma unroll
  for (int nt = 0; nt < 8; ++nt)
#pragma unroll
    for (int i = 0; i < 8; ++i) outacc[nt][i] = 0.0f;

  for (int qb = 0; qb < 4; ++qb) {
    // ================= GEMM1 chunk =================
    // G[t][c] = zb . F1 (+bias): M=16 tok, N=1024, K=1024
    {
      const __bf16* Arow = ZB + lane_lo * 1024;
      const __bf16* Bbase =
          F1bT + (size_t)(qb * 1024 + wave * 128 + lane_lo) * 1024;
      v8f g1[8];
#pragma unroll
      for (int nt = 0; nt < 8; ++nt) {
        const float bias = Bias1[qb * 1024 + wave * 128 + nt * 16 + lane_lo];
#pragma unroll
        for (int i = 0; i < 8; ++i) g1[nt][i] = bias;
      }
      v16bf b[3][8];
#pragma unroll
      for (int nt = 0; nt < 8; ++nt)
        b[0][nt] = fragB(Bbase, nt * 16 * 1024, lane_hi);
#pragma unroll
      for (int nt = 0; nt < 8; ++nt)
        b[1][nt] = fragB(Bbase, nt * 16 * 1024 + 32, lane_hi);
#pragma unroll
      for (int kk = 0; kk < 32; ++kk) {
        const v16bf a = fragA(Arow, kk * 32, lane_hi);
        if (kk < 30) {
#pragma unroll
          for (int nt = 0; nt < 8; ++nt)
            b[(kk + 2) % 3][nt] =
                fragB(Bbase, nt * 16 * 1024 + (kk + 2) * 32, lane_hi);
        }
#pragma unroll
        for (int nt = 0; nt < 8; ++nt)
          g1[nt] = wmma_bf16(a, b[kk % 3][nt], g1[nt]);
        __builtin_amdgcn_sched_group_barrier(SG_DSRD, 2, 0);
        __builtin_amdgcn_sched_group_barrier(SG_VMEMRD, 16, 0);
        __builtin_amdgcn_sched_group_barrier(SG_WMMA, 8, 0);
      }
      // store D tiles -> G_s (bf16)
#pragma unroll
      for (int nt = 0; nt < 8; ++nt) {
        const int cc = (wave * 8 + nt) * 16 + lane_lo;
#pragma unroll
        for (int i = 0; i < 8; ++i)
          G_s[(lane_hi * 8 + i) * 1024 + cc] = (__bf16)g1[nt][i];
      }
    }
    __syncthreads();

    // ================= middle: M[t][r][q] = A_t . G_t =================
    // per-token 64x16, K=64; G fragments hoisted across the 4 r-tiles
#pragma unroll
    for (int u = 0; u < 2; ++u) {
      const int tok = wave * 2 + u;
      const __bf16* Gr = G_s + tok * 1024 + lane_lo * 64;
      const v16bf bg0 = fragB(Gr, 0, lane_hi);
      const v16bf bg1 = fragB(Gr, 32, lane_hi);
      const __bf16* Abase = A_s + tok * 4096;
#pragma unroll
      for (int rt = 0; rt < 4; ++rt) {
        const __bf16* Ar = Abase + (rt * 16 + lane_lo) * 64;
        const v16bf a0 = fragA(Ar, 0, lane_hi);
        const v16bf a1v = fragA(Ar, 32, lane_hi);
        v8f acc;
#pragma unroll
        for (int i = 0; i < 8; ++i) acc[i] = 0.0f;
        acc = wmma_bf16(a0, bg0, acc);
        acc = wmma_bf16(a1v, bg1, acc);
        const int q = lane_lo;
#pragma unroll
        for (int i = 0; i < 8; ++i)
          M_s[tok * 1024 + q * 64 + rt * 16 + lane_hi * 8 + i] = (__bf16)acc[i];
      }
    }
    __syncthreads();

    // ================= GEMM2 partial =================
    // out[t][o] += M[t][.] . F2b[o][.], K-chunk = 1024; same pipelined shape
    {
      const __bf16* Arow = M_s + lane_lo * 1024;
      const __bf16* Bbase =
          F2b + (size_t)(wave * 128 + lane_lo) * 4096 + qb * 1024;
      v16bf b[3][8];
#pragma unroll
      for (int nt = 0; nt < 8; ++nt)
        b[0][nt] = fragB(Bbase, nt * 16 * 4096, lane_hi);
#pragma unroll
      for (int nt = 0; nt < 8; ++nt)
        b[1][nt] = fragB(Bbase, nt * 16 * 4096 + 32, lane_hi);
#pragma unroll
      for (int kk = 0; kk < 32; ++kk) {
        const v16bf a = fragA(Arow, kk * 32, lane_hi);
        if (kk < 30) {
#pragma unroll
          for (int nt = 0; nt < 8; ++nt)
            b[(kk + 2) % 3][nt] =
                fragB(Bbase, nt * 16 * 4096 + (kk + 2) * 32, lane_hi);
        }
#pragma unroll
        for (int nt = 0; nt < 8; ++nt)
          outacc[nt] = wmma_bf16(a, b[kk % 3][nt], outacc[nt]);
        __builtin_amdgcn_sched_group_barrier(SG_DSRD, 2, 0);
        __builtin_amdgcn_sched_group_barrier(SG_VMEMRD, 16, 0);
        __builtin_amdgcn_sched_group_barrier(SG_WMMA, 8, 0);
      }
    }
    __syncthreads();
  }

  // ---- store output tile (f32) ----
#pragma unroll
  for (int nt = 0; nt < 8; ++nt) {
    const int o = wave * 128 + nt * 16 + lane_lo;
#pragma unroll
    for (int i = 0; i < 8; ++i)
      out[(size_t)(tok0 + lane_hi * 8 + i) * ODIM + o] = outacc[nt][i];
  }
}

// =====================================================================
extern "C" void kernel_launch(void* const* d_in, const int* in_sizes, int n_in,
                              void* d_out, int out_size, void* d_ws, size_t ws_size,
                              hipStream_t stream) {
  const float* z       = (const float*)d_in[0];   // [8,2048,1024]
  const float* proj0   = (const float*)d_in[1];   // [1024,32]
  const float* factor0 = (const float*)d_in[2];   // [64,32,64]
  const float* factor1 = (const float*)d_in[3];   // [64,1025,64]
  const float* factor2 = (const float*)d_in[4];   // [64,1024,64]
  float* out = (float*)d_out;

  char* ws = (char*)d_ws;
  float*  bn    = (float*)(ws + OFF_BN);
  float*  h     = (float*)(ws + OFF_H);
  float*  a1f   = (float*)(ws + OFF_A1);
  __bf16* F1bT  = (__bf16*)(ws + OFF_F1);
  float*  Bias1 = (float*)(ws + OFF_B1);
  __bf16* F2b   = (__bf16*)(ws + OFF_F2);

  (void)in_sizes; (void)n_in; (void)out_size; (void)ws_size;

  hipFuncSetAttribute(reinterpret_cast<const void*>(tt_fused_kernel),
                      hipFuncAttributeMaxDynamicSharedMemorySize,
                      (int)SMEM_BYTES);

  zero_bn_kernel<<<1, 64, 0, stream>>>(bn);
  conv_f1_kernel<<<4096, 256, 0, stream>>>(factor1, F1bT, Bias1);
  conv_f2_kernel<<<1024, 256, 0, stream>>>(factor2, F2b);
  gating_kernel<<<NTOK / 8, 256, 0, stream>>>(z, proj0, h, bn);
  entmax_kernel<<<NTOK / 256, 256, 0, stream>>>(h, bn, a1f);
  tt_fused_kernel<<<NTOK / MTILE, 256, SMEM_BYTES, stream>>>(
      z, a1f, factor0, F1bT, Bias1, F2b, out);
}
